// FluxIPAttnProcessor_54107997995367
// MI455X (gfx1250) — matrix operand order repaired and checked by reading, not compiled, not run on hardware
//
#include <hip/hip_runtime.h>

// ---------------------------------------------------------------------------
// Types
// ---------------------------------------------------------------------------
typedef __attribute__((ext_vector_type(16))) __bf16    bf16x16;
typedef __attribute__((ext_vector_type(8)))  float     f32x8;
typedef __attribute__((ext_vector_type(4)))  float     f32x4;
typedef __attribute__((ext_vector_type(4)))  unsigned  u32x4;
typedef __attribute__((ext_vector_type(2)))  unsigned  u32x2;

union Frag {
    unsigned int u[8];
    u32x4        q[2];
    bf16x16      v;
};

// Hardware packed f32->bf16 conversion (1 VALU op for 2 elements, RNE).
__device__ __forceinline__ unsigned int pack2_bf16(float a, float b) {
    unsigned int r;
    asm("v_cvt_pk_bf16_f32 %0, %1, %2" : "=v"(r) : "v"(a), "v"(b));
    return r;
}
__device__ __forceinline__ unsigned short bf16_1(float a) {
    return (unsigned short)(pack2_bf16(a, a) & 0xFFFFu);
}

// ---------------------------------------------------------------------------
// Generic GEMM:  C[M,N] = A[M,K] @ W[N,K]^T + bias[N]   (fp32 in, fp32 out)
// Block: 256 threads (8 waves). Tile: 128(M) x 128(N), K staged 64 at a time.
// Double-buffered LDS, software pipelined: global loads of stage s overlap
// the 16-WMMA compute of stage s-1; one barrier per stage.
// ---------------------------------------------------------------------------
#define ASTR 72   // LDS row stride (bf16 elems): 64 + 8 pad (144B, 16B mult)

__device__ __forceinline__ void cvt_store_tile(unsigned short* base, int st_row,
                                               int st_kk, const f32x4* f) {
    unsigned int* dst = (unsigned int*)(base + st_row * ASTR + st_kk);
    #pragma unroll
    for (int j = 0; j < 4; ++j) {
        u32x4 s;
        s[0] = pack2_bf16(f[2 * j][0],     f[2 * j][1]);
        s[1] = pack2_bf16(f[2 * j][2],     f[2 * j][3]);
        s[2] = pack2_bf16(f[2 * j + 1][0], f[2 * j + 1][1]);
        s[3] = pack2_bf16(f[2 * j + 1][2], f[2 * j + 1][3]);
        *(u32x4*)(dst + 4 * j) = s;
    }
}

__device__ __forceinline__ void compute_stage(const unsigned short* Asb,
                                              const unsigned short* Wsb,
                                              int wave, int mloc, int m_half,
                                              f32x8* acc) {
    #pragma unroll
    for (int ks = 0; ks < 2; ++ks) {
        // A fragment (16x32, A layout: half0 K {0..7,16..23}, half1 +8)
        Frag af;
        {
            const unsigned short* p =
                Asb + (wave * 16 + mloc) * ASTR + ks * 32 + m_half * 8;
            af.q[0] = *(const u32x4*)p;
            af.q[1] = *(const u32x4*)(p + 16);
        }
        #pragma unroll
        for (int g = 0; g < 8; ++g) {
            Frag bfg;
            const unsigned short* p =
                Wsb + (g * 16 + mloc) * ASTR + ks * 32 + m_half * 8;
            bfg.q[0] = *(const u32x4*)p;
            bfg.q[1] = *(const u32x4*)(p + 16);
            acc[g] = __builtin_amdgcn_wmma_f32_16x16x32_bf16(
                false, af.v, false, bfg.v, (short)0, acc[g], false, false);
        }
    }
}

__global__ __launch_bounds__(256)
void gemm_bias_kernel(const float* __restrict__ A, const float* __restrict__ W,
                      const float* __restrict__ bias, float* __restrict__ C,
                      int M, int N, int K)
{
    __shared__ __align__(16) unsigned short As[2][128 * ASTR];  // 2 x 18.4 KB
    __shared__ __align__(16) unsigned short Ws[2][128 * ASTR];  // 2 x 18.4 KB

    const int tid    = threadIdx.x;
    const int wave   = tid >> 5;
    const int lane   = tid & 31;
    const int m_half = lane >> 4;     // 0: lanes 0-15, 1: lanes 16-31
    const int mloc   = lane & 15;
    const int bm     = blockIdx.y * 128;
    const int bn     = blockIdx.x * 128;

    const int st_row = tid >> 1, st_kk = (tid & 1) * 32;
    const float* a_src = A + (size_t)(bm + st_row) * K + st_kk;
    const float* w_src = W + (size_t)(bn + st_row) * K + st_kk;

    f32x8 acc[8];
    #pragma unroll
    for (int g = 0; g < 8; ++g)
        #pragma unroll
        for (int r = 0; r < 8; ++r) acc[g][r] = 0.0f;

    const int nstage = K >> 6;    // K / 64 (K is a multiple of 64 here)

    // ---- prologue: stage 0 -> buffer 0
    f32x4 fa[8], fw[8];
    #pragma unroll
    for (int i = 0; i < 8; ++i) fa[i] = *(const f32x4*)(a_src + 4 * i);
    #pragma unroll
    for (int i = 0; i < 8; ++i) fw[i] = *(const f32x4*)(w_src + 4 * i);
    cvt_store_tile(As[0], st_row, st_kk, fa);
    cvt_store_tile(Ws[0], st_row, st_kk, fw);

    // ---- pipelined main loop
    for (int s = 1; s <= nstage; ++s) {
        const int k0 = s << 6;
        if (s < nstage) {
            #pragma unroll
            for (int i = 0; i < 8; ++i) fa[i] = *(const f32x4*)(a_src + k0 + 4 * i);
            #pragma unroll
            for (int i = 0; i < 8; ++i) fw[i] = *(const f32x4*)(w_src + k0 + 4 * i);
            if (s + 1 < nstage) {
                __builtin_prefetch(a_src + k0 + 64, 0, 1);
                __builtin_prefetch(w_src + k0 + 64, 0, 1);
            }
        }
        __syncthreads();
        compute_stage(As[(s - 1) & 1], Ws[(s - 1) & 1], wave, mloc, m_half, acc);
        if (s < nstage) {
            cvt_store_tile(As[s & 1], st_row, st_kk, fa);
            cvt_store_tile(Ws[s & 1], st_row, st_kk, fw);
        }
    }

    // ---- epilogue: C layout row = r + 8*m_half (+wave*16), col = g*16+mloc
    #pragma unroll
    for (int g = 0; g < 8; ++g) {
        const int col = bn + g * 16 + mloc;
        const float b = bias ? bias[col] : 0.0f;
        #pragma unroll
        for (int r = 0; r < 8; ++r) {
            const int row = bm + wave * 16 + r + 8 * m_half;
            C[(size_t)row * N + col] = acc[g][r] + b;
        }
    }
}

// ---------------------------------------------------------------------------
// Row op: optional RMSNorm(DH=128) + optional RoPE, fp32 -> bf16 relayout
// into [H=24][S_out_total][128].  One wave per (s,h) row, 4 elems per lane.
// ---------------------------------------------------------------------------
__global__ __launch_bounds__(256)
void rowop_kernel(const float* __restrict__ in, const float* __restrict__ normw,
                  const float* __restrict__ rc, const float* __restrict__ rs,
                  unsigned short* __restrict__ out,
                  int S_in, int s_out_off, int S_out_total)
{
    const int wave = threadIdx.x >> 5;
    const int lane = threadIdx.x & 31;
    const int row  = blockIdx.x * 8 + wave;
    if (row >= S_in * 24) return;
    const int s  = row % S_in;
    const int hh = row / S_in;
    const int d  = lane * 4;

    const float* px = in + (size_t)s * 3072 + hh * 128 + d;
    float x0 = px[0], x1 = px[1], x2 = px[2], x3 = px[3];

    if (normw) {
        float ss = x0 * x0 + x1 * x1 + x2 * x2 + x3 * x3;
        ss += __shfl_xor(ss, 16, 32);
        ss += __shfl_xor(ss, 8, 32);
        ss += __shfl_xor(ss, 4, 32);
        ss += __shfl_xor(ss, 2, 32);
        ss += __shfl_xor(ss, 1, 32);
        const float rinv = rsqrtf(ss * (1.0f / 128.0f) + 1e-6f);
        x0 *= rinv * normw[d + 0];
        x1 *= rinv * normw[d + 1];
        x2 *= rinv * normw[d + 2];
        x3 *= rinv * normw[d + 3];
    }
    if (rc) {
        const int pos = s_out_off + s;
        const float* pc = rc + (size_t)pos * 128 + d;
        const float* ps = rs + (size_t)pos * 128 + d;
        float c0 = pc[0], c1 = pc[1], c2 = pc[2], c3 = pc[3];
        float t0 = ps[0], t1 = ps[1], t2 = ps[2], t3 = ps[3];
        float y0 = x0 * c0 - x1 * t0;
        float y1 = x1 * c1 + x0 * t1;
        float y2 = x2 * c2 - x3 * t2;
        float y3 = x3 * c3 + x2 * t3;
        x0 = y0; x1 = y1; x2 = y2; x3 = y3;
    }
    u32x2 pk;
    pk[0] = pack2_bf16(x0, x1);
    pk[1] = pack2_bf16(x2, x3);
    *(u32x2*)(out + ((size_t)hh * S_out_total + s_out_off + s) * 128 + d) = pk;
}

// ---------------------------------------------------------------------------
// Flash-style WMMA attention.  Q,K,V are bf16 [H][S][128].
// Block = 128 threads (4 waves); wave owns 16 query rows; block covers 64.
// Key loop: 32 keys/iter.  QK^T: 8 WMMAs; online softmax; PV: 8 WMMAs.
// Output fp32 merged layout [Sq][3072] at col h*128+d.
// ---------------------------------------------------------------------------
#define KSTR 136  // LDS K-tile row stride (bf16): 128 + 8 pad
#define VSTR 40   // LDS V^T row stride  (bf16): 32 + 8 pad (16B aligned rows)

__global__ __launch_bounds__(128)
void attn_kernel(const unsigned short* __restrict__ Q,
                 const unsigned short* __restrict__ Km,
                 const unsigned short* __restrict__ Vm,
                 float* __restrict__ Out,
                 int Sq, int Sk, float scale)
{
    __shared__ __align__(16) unsigned short kt[32 * KSTR];    // keys x dh
    __shared__ __align__(16) unsigned short vt[128 * VSTR];   // dh x keys
    __shared__ __align__(16) unsigned short pt[4][16 * 32];   // per wave P

    const int tid    = threadIdx.x;
    const int wave   = tid >> 5;
    const int lane   = tid & 31;
    const int m_half = lane >> 4;
    const int mloc   = lane & 15;
    const int h      = blockIdx.y;
    const int q0     = blockIdx.x * 64 + wave * 16;

    // Q fragments (16x128 => 4 chunks of K=32), A layout, straight from global
    Frag qf[4];
    {
        const unsigned short* qp = Q + ((size_t)h * Sq + q0 + mloc) * 128;
        #pragma unroll
        for (int c = 0; c < 4; ++c) {
            const unsigned short* p = qp + c * 32 + m_half * 8;
            qf[c].q[0] = *(const u32x4*)p;
            qf[c].q[1] = *(const u32x4*)(p + 16);
        }
    }

    f32x8 o[8];
    float mrow[8], lrow[8];
    #pragma unroll
    for (int g = 0; g < 8; ++g)
        #pragma unroll
        for (int r = 0; r < 8; ++r) o[g][r] = 0.0f;
    #pragma unroll
    for (int r = 0; r < 8; ++r) { mrow[r] = -3.0e38f; lrow[r] = 0.0f; }

    const int ld_row = tid >> 2;          // key 0..31
    const int ld_off = (tid & 3) * 32;    // dh offset 0/32/64/96

    for (int j0 = 0; j0 < Sk; j0 += 32) {
        // ---- cooperative staging: batch all global loads into regs first
        {
            const unsigned short* ks = Km + ((size_t)h * Sk + j0 + ld_row) * 128 + ld_off;
            const unsigned short* vs = Vm + ((size_t)h * Sk + j0 + ld_row) * 128 + ld_off;
            u32x4 kv[4], vv[4];
            #pragma unroll
            for (int i = 0; i < 4; ++i) kv[i] = *(const u32x4*)(ks + 8 * i);
            #pragma unroll
            for (int i = 0; i < 4; ++i) vv[i] = *(const u32x4*)(vs + 8 * i);
            // K tile row-major
            u32x4* kd = (u32x4*)&kt[ld_row * KSTR + ld_off];
            #pragma unroll
            for (int i = 0; i < 4; ++i) kd[i] = kv[i];
            // V tile transposed: extract 16-bit halves from registers
            #pragma unroll
            for (int b = 0; b < 4; ++b)
                #pragma unroll
                for (int i = 0; i < 4; ++i) {
                    const unsigned int w32 = vv[b][i];
                    const int dd = ld_off + b * 8 + i * 2;
                    vt[dd * VSTR + ld_row]       = (unsigned short)(w32 & 0xFFFFu);
                    vt[(dd + 1) * VSTR + ld_row] = (unsigned short)(w32 >> 16);
                }
        }
        __syncthreads();

        // ---- scores: two 16x16 tiles (keys j0..j0+15, j0+16..j0+31)
        f32x8 s0, s1;
        #pragma unroll
        for (int r = 0; r < 8; ++r) { s0[r] = 0.0f; s1[r] = 0.0f; }
        #pragma unroll
        for (int c = 0; c < 4; ++c) {
            Frag b0, b1;
            const unsigned short* p0 = &kt[mloc * KSTR + c * 32 + m_half * 8];
            b0.q[0] = *(const u32x4*)p0;
            b0.q[1] = *(const u32x4*)(p0 + 16);
            const unsigned short* p1 = &kt[(16 + mloc) * KSTR + c * 32 + m_half * 8];
            b1.q[0] = *(const u32x4*)p1;
            b1.q[1] = *(const u32x4*)(p1 + 16);
            s0 = __builtin_amdgcn_wmma_f32_16x16x32_bf16(
                false, qf[c].v, false, b0.v, (short)0, s0, false, false);
            s1 = __builtin_amdgcn_wmma_f32_16x16x32_bf16(
                false, qf[c].v, false, b1.v, (short)0, s1, false, false);
        }

        // ---- online softmax (row = r + 8*m_half; 16 cols across half-wave)
        #pragma unroll
        for (int r = 0; r < 8; ++r) {
            const float a = s0[r] * scale;
            const float b = s1[r] * scale;
            float v = fmaxf(a, b);
            v = fmaxf(v, __shfl_xor(v, 1, 32));
            v = fmaxf(v, __shfl_xor(v, 2, 32));
            v = fmaxf(v, __shfl_xor(v, 4, 32));
            v = fmaxf(v, __shfl_xor(v, 8, 32));
            const float nm = fmaxf(mrow[r], v);
            const float f  = __expf(mrow[r] - nm);
            mrow[r] = nm;
            const float p0 = __expf(a - nm);
            const float p1 = __expf(b - nm);
            float ps = p0 + p1;
            ps += __shfl_xor(ps, 1, 32);
            ps += __shfl_xor(ps, 2, 32);
            ps += __shfl_xor(ps, 4, 32);
            ps += __shfl_xor(ps, 8, 32);
            lrow[r] = lrow[r] * f + ps;
            #pragma unroll
            for (int g = 0; g < 8; ++g) o[g][r] *= f;
            const int rr = r + 8 * m_half;
            pt[wave][rr * 32 + mloc]      = bf16_1(p0);
            pt[wave][rr * 32 + 16 + mloc] = bf16_1(p1);
        }
        // same-wave LDS RAW: force DS completion before A-layout readback
        asm volatile("s_wait_dscnt 0x0" ::: "memory");

        // ---- P fragment (16x32, A layout) from LDS
        Frag pf;
        {
            const unsigned short* p = &pt[wave][mloc * 32 + m_half * 8];
            pf.q[0] = *(const u32x4*)p;
            pf.q[1] = *(const u32x4*)(p + 16);
        }
        // ---- O += P @ V  (8 dh-column tiles of 16)
        #pragma unroll
        for (int g = 0; g < 8; ++g) {
            Frag vfg;
            const unsigned short* p = &vt[(g * 16 + mloc) * VSTR + m_half * 8];
            vfg.q[0] = *(const u32x4*)p;
            vfg.q[1] = *(const u32x4*)(p + 16);
            o[g] = __builtin_amdgcn_wmma_f32_16x16x32_bf16(
                false, pf.v, false, vfg.v, (short)0, o[g], false, false);
        }
        __syncthreads();
    }

    // ---- write O / l to merged fp32 [Sq][3072]
    #pragma unroll
    for (int g = 0; g < 8; ++g) {
        const int col = h * 128 + g * 16 + mloc;
        #pragma unroll
        for (int r = 0; r < 8; ++r) {
            const int row = q0 + r + 8 * m_half;
            Out[(size_t)row * 3072 + col] = o[g][r] / lrow[r];
        }
    }
}

// ---------------------------------------------------------------------------
// img += ip_out * mask   (bbox (8,8,24,24) on 32x32 grid, IP_SCALE = 1.0)
// ---------------------------------------------------------------------------
__global__ __launch_bounds__(256)
void maskadd_kernel(float* __restrict__ img, const float* __restrict__ ip)
{
    const int idx = blockIdx.x * 256 + threadIdx.x;
    const int s = idx / 3072;
    const int y = s >> 5, x = s & 31;
    if (y >= 8 && y < 24 && x >= 8 && x < 24) img[idx] += ip[idx];
}

// ---------------------------------------------------------------------------
// Host launch
// ---------------------------------------------------------------------------
extern "C" void kernel_launch(void* const* d_in, const int* in_sizes, int n_in,
                              void* d_out, int out_size, void* d_ws, size_t ws_size,
                              hipStream_t stream)
{
    const float* hidden = (const float*)d_in[0];
    const float* enc    = (const float*)d_in[1];
    const float* iph    = (const float*)d_in[2];
    const float* rc     = (const float*)d_in[3];
    const float* rs     = (const float*)d_in[4];
    const float* Wq  = (const float*)d_in[5];  const float* bq  = (const float*)d_in[6];
    const float* Wk  = (const float*)d_in[7];  const float* bk  = (const float*)d_in[8];
    const float* Wv  = (const float*)d_in[9];  const float* bv  = (const float*)d_in[10];
    const float* nq  = (const float*)d_in[11]; const float* nk  = (const float*)d_in[12];
    const float* Wqa = (const float*)d_in[13]; const float* bqa = (const float*)d_in[14];
    const float* Wka = (const float*)d_in[15]; const float* bka = (const float*)d_in[16];
    const float* Wva = (const float*)d_in[17]; const float* bva = (const float*)d_in[18];
    const float* nqa = (const float*)d_in[19]; const float* nka = (const float*)d_in[20];
    const float* Wo  = (const float*)d_in[21]; const float* bo  = (const float*)d_in[22];
    const float* Wao = (const float*)d_in[23]; const float* bao = (const float*)d_in[24];
    const float* Wkip = (const float*)d_in[25]; const float* bkip = (const float*)d_in[26];
    const float* Wvip = (const float*)d_in[27]; const float* bvip = (const float*)d_in[28];
    const float* nipq = (const float*)d_in[29]; const float* nipk = (const float*)d_in[30];

    float* outp = (float*)d_out;

    // workspace carve-up
    char* w = (char*)d_ws;
    auto alloc_f = [&](size_t n) { float* p = (float*)w; w += n * 4; return p; };
    float* q_lin   = alloc_f(1024ull * 3072);
    float* k_lin   = alloc_f(1024ull * 3072);
    float* v_lin   = alloc_f(1024ull * 3072);
    float* eq_lin  = alloc_f(512ull * 3072);
    float* ek_lin  = alloc_f(512ull * 3072);
    float* ev_lin  = alloc_f(512ull * 3072);
    float* ipk_lin = alloc_f(128ull * 3072);
    float* ipv_lin = alloc_f(128ull * 3072);
    float* attn_o  = alloc_f(1536ull * 3072);
    float* ip_attn = alloc_f(1024ull * 3072);
    auto alloc_h = [&](size_t n) { unsigned short* p = (unsigned short*)w; w += n * 2; return p; };
    unsigned short* qf   = alloc_h(24ull * 1536 * 128);
    unsigned short* kf   = alloc_h(24ull * 1536 * 128);
    unsigned short* vf   = alloc_h(24ull * 1536 * 128);
    unsigned short* ipq  = alloc_h(24ull * 1024 * 128);
    unsigned short* ipkh = alloc_h(24ull * 128 * 128);
    unsigned short* ipvh = alloc_h(24ull * 128 * 128);

    const dim3 blk(256);

    // QKV projections (hidden: M=1024, encoder: M=512, ip: M=128 / K=1152)
    gemm_bias_kernel<<<dim3(24, 8), blk, 0, stream>>>(hidden, Wq,  bq,  q_lin,   1024, 3072, 3072);
    gemm_bias_kernel<<<dim3(24, 8), blk, 0, stream>>>(hidden, Wk,  bk,  k_lin,   1024, 3072, 3072);
    gemm_bias_kernel<<<dim3(24, 8), blk, 0, stream>>>(hidden, Wv,  bv,  v_lin,   1024, 3072, 3072);
    gemm_bias_kernel<<<dim3(24, 4), blk, 0, stream>>>(enc,    Wqa, bqa, eq_lin,  512,  3072, 3072);
    gemm_bias_kernel<<<dim3(24, 4), blk, 0, stream>>>(enc,    Wka, bka, ek_lin,  512,  3072, 3072);
    gemm_bias_kernel<<<dim3(24, 4), blk, 0, stream>>>(enc,    Wva, bva, ev_lin,  512,  3072, 3072);
    gemm_bias_kernel<<<dim3(24, 1), blk, 0, stream>>>(iph,    Wkip, bkip, ipk_lin, 128, 3072, 1152);
    gemm_bias_kernel<<<dim3(24, 1), blk, 0, stream>>>(iph,    Wvip, bvip, ipv_lin, 128, 3072, 1152);

    // head relayout + RMS (+ RoPE for main q/k, concat order [txt, img])
    rowop_kernel<<<dim3(1024*24/8), blk, 0, stream>>>(q_lin,  nipq, nullptr, nullptr, ipq,  1024, 0,   1024);
    rowop_kernel<<<dim3(128*24/8),  blk, 0, stream>>>(ipk_lin, nipk, nullptr, nullptr, ipkh, 128,  0,   128);
    rowop_kernel<<<dim3(128*24/8),  blk, 0, stream>>>(ipv_lin, nullptr, nullptr, nullptr, ipvh, 128, 0, 128);
    rowop_kernel<<<dim3(512*24/8),  blk, 0, stream>>>(eq_lin, nqa, rc, rs, qf, 512,  0,   1536);
    rowop_kernel<<<dim3(1024*24/8), blk, 0, stream>>>(q_lin,  nq,  rc, rs, qf, 1024, 512, 1536);
    rowop_kernel<<<dim3(512*24/8),  blk, 0, stream>>>(ek_lin, nka, rc, rs, kf, 512,  0,   1536);
    rowop_kernel<<<dim3(1024*24/8), blk, 0, stream>>>(k_lin,  nk,  rc, rs, kf, 1024, 512, 1536);
    rowop_kernel<<<dim3(512*24/8),  blk, 0, stream>>>(ev_lin, nullptr, nullptr, nullptr, vf, 512,  0,   1536);
    rowop_kernel<<<dim3(1024*24/8), blk, 0, stream>>>(v_lin,  nullptr, nullptr, nullptr, vf, 1024, 512, 1536);

    const float scale = 0.08838834764831845f;   // 1/sqrt(128)
    attn_kernel<<<dim3(16, 24), dim3(128), 0, stream>>>(ipq, ipkh, ipvh, ip_attn, 1024, 128,  scale);
    attn_kernel<<<dim3(24, 24), dim3(128), 0, stream>>>(qf,  kf,   vf,   attn_o,  1536, 1536, scale);

    maskadd_kernel<<<dim3(1024*3072/256), blk, 0, stream>>>(attn_o + 512ull*3072, ip_attn);

    // output projections straight into d_out: img (1024x3072) then enc (512x3072)
    gemm_bias_kernel<<<dim3(24, 8), blk, 0, stream>>>(attn_o + 512ull*3072, Wo,  bo,  outp,                1024, 3072, 3072);
    gemm_bias_kernel<<<dim3(24, 4), blk, 0, stream>>>(attn_o,               Wao, bao, outp + 1024ull*3072, 512,  3072, 3072);
}
